// MultiHeadAttention_18511309046043
// MI455X (gfx1250) — matrix operand order repaired and checked by reading, not compiled, not run on hardware
//
#include <hip/hip_runtime.h>
#include <hip/hip_fp16.h>
#include <stdint.h>

typedef __attribute__((ext_vector_type(16))) _Float16 v16h;
typedef __attribute__((ext_vector_type(8)))  _Float16 v8h;
typedef __attribute__((ext_vector_type(4)))  _Float16 v4h;
typedef __attribute__((ext_vector_type(8)))  float    v8f;
typedef __attribute__((ext_vector_type(4)))  uint32_t v4u;
typedef __attribute__((ext_vector_type(8)))  uint32_t v8u;

#define BB 4
#define TT 2048
#define CC 1024
#define NH 16
#define HD 64
#define MROWS 8192   // B*T: rows of every activation matrix

static __device__ __forceinline__ v16h cat8(v8h lo, v8h hi) {
    v16h r;
#pragma unroll
    for (int i = 0; i < 8; ++i) { r[i] = lo[i]; r[i + 8] = hi[i]; }
    return r;
}

static __device__ __forceinline__ v8f wmma_f16(v16h a, v16h b, v8f c) {
    return __builtin_amdgcn_wmma_f32_16x16x32_f16(
        /*neg_a=*/false, a, /*neg_b=*/false, b,
        /*c_mod=*/(short)0, c, /*reuse_a=*/false, /*reuse_b=*/false);
}

// ---------------------------------------------------------------------------
// fp32 -> f16 bulk conversion (vectorized by 4)
// ---------------------------------------------------------------------------
__global__ void __launch_bounds__(256)
cvt_f16(const float* __restrict__ in, _Float16* __restrict__ out, int n4) {
    const int i = blockIdx.x * 256 + threadIdx.x;
    if (i < n4) {
        float4 f = ((const float4*)in)[i];
        v4h h;
        h[0] = (_Float16)f.x; h[1] = (_Float16)f.y;
        h[2] = (_Float16)f.z; h[3] = (_Float16)f.w;
        ((v4h*)out)[i] = h;
    }
}

// ---------------------------------------------------------------------------
// Tiled GEMM: out(8192,1024) = A(8192,1024) @ W(1024,1024)^T + bias (A,W f16)
//   mode: 0 -> f16 out (b,h,t,hd)   1 -> f16 out (b,h,hd,t)   2 -> f32 out
// Block: 256 threads = 8 waves, tile 128x128, K staged 32 at a time.
// A tile is fetched by the Tensor Data Mover (tensor_load_to_lds, D# built
// per ISA 8.3/8.4 with LDS padding), W tile by global_load_async_to_lds_b128.
// ---------------------------------------------------------------------------
__global__ void __launch_bounds__(256)
gemm128_wmma(const _Float16* __restrict__ A, const _Float16* __restrict__ Wh,
             const float* __restrict__ bias, void* __restrict__ outv, int mode) {
    __shared__ __align__(16) _Float16 As[128][40];  // 80B rows (64B data + 16B pad)
    __shared__ __align__(16) _Float16 Ws[128][40];

    const int tid  = threadIdx.x;
    const int lane = tid & 31;
    const int wid  = tid >> 5;
    const int lm   = lane & 15;
    const int lh   = lane >> 4;
    const int m0   = blockIdx.x * 128;
    const int n0   = blockIdx.y * 128;
    const int row  = tid >> 1;    // staging row (W tile)
    const int hsel = tid & 1;     // which 16-half chunk of the row

    const uint32_t as_lds = (uint32_t)(uintptr_t)&As[0][0];            // LDS byte addr
    const uint32_t ws_dst = (uint32_t)(uintptr_t)&Ws[row][hsel * 16];  // per-lane LDS dest
    const _Float16* wsrc  = Wh + (size_t)(n0 + row) * CC + hsel * 16;

    v8f acc[8] = {};

    for (int k0 = 0; k0 < CC; k0 += 32) {
        // ---- A tile (128x32) via TDM: one descriptor, issued by wave 0 ----
        if (wid == 0) {
            const uint64_t ga = (uint64_t)(uintptr_t)(A + (size_t)m0 * CC + k0);
            v4u g0;
            g0[0] = 1u;                                   // count=1 (valid), no gather
            g0[1] = as_lds;                               // lds_addr
            g0[2] = (uint32_t)ga;                         // global_addr[31:0]
            g0[3] = ((uint32_t)(ga >> 32) & 0x01FFFFFFu)  // global_addr[56:32]
                    | 0x80000000u;                        // type=2 ("image")
            v8u g1;
            g1[0] = 0x06D10000u;            // data_size=2B, pad_en, every 16DW pad 4DW
            g1[1] = (uint32_t)CC << 16;     // tensor_dim0 = 1024 (lo16)
            g1[2] = ((uint32_t)MROWS & 0xFFFFu) << 16;         // tensor_dim1 lo16
            g1[3] = ((uint32_t)MROWS >> 16) | (32u << 16);     // dim1 hi16 | tile_dim0=32
            g1[4] = 128u;                   // tile_dim1 = 128 rows
            g1[5] = (uint32_t)CC;           // tensor_dim0_stride = 1024
            g1[6] = 0u;
            g1[7] = 0u;
            asm volatile("tensor_load_to_lds %0, %1" :: "s"(g0), "s"(g1) : "memory");
        }
        // ---- W tile (128x32) via async global->LDS copies (all threads) ----
        {
            const uint64_t ga = (uint64_t)(uintptr_t)(wsrc + k0);
            asm volatile("global_load_async_to_lds_b128 %0, %1, off"
                         :: "v"(ws_dst), "v"(ga) : "memory");
            asm volatile("global_load_async_to_lds_b128 %0, %1, off offset:16"
                         :: "v"(ws_dst), "v"(ga) : "memory");
        }
        asm volatile("s_wait_asynccnt 0" ::: "memory");
        if (wid == 0) __builtin_amdgcn_s_wait_tensorcnt(0);
        __syncthreads();

        // ---- each wave computes its 16x128 strip: 8 WMMAs per chunk ----
        const int ar  = wid * 16 + lm;
        const int klo = lh * 8;
        v16h af = cat8(*(const v8h*)&As[ar][klo],
                       *(const v8h*)&As[ar][16 + klo]);
        const int koff = lh * 16;
#pragma unroll
        for (int nt = 0; nt < 8; ++nt) {
            const int br = nt * 16 + lm;
            v16h bf = cat8(*(const v8h*)&Ws[br][klo],
                           *(const v8h*)&Ws[br][16 + klo]);
            acc[nt] = wmma_f16(af, bf, acc[nt]);
        }
        __syncthreads();
    }

    // ---- epilogue: lane holds col n=lane%16, rows r+8*(lane/16) ----
    const int mbase = m0 + wid * 16 + lh * 8;
    const int nbase = n0 + lm;
#pragma unroll
    for (int nt = 0; nt < 8; ++nt) {
        const int n = nbase + nt * 16;
        const float bv = bias[n];
#pragma unroll
        for (int r = 0; r < 8; ++r) {
            const int m = mbase + r;
            const float val = acc[nt][r] + bv;
            if (mode == 2) {
                ((float*)outv)[(size_t)m * CC + n] = val;
            } else {
                const int b = m >> 11, t = m & (TT - 1);
                const int h = n >> 6,  hd = n & (HD - 1);
                size_t idx;
                if (mode == 0)
                    idx = ((size_t)((b * NH + h) * TT + t)) * HD + hd;
                else
                    idx = ((size_t)((b * NH + h) * HD + hd)) * TT + t;
                ((_Float16*)outv)[idx] = (_Float16)val;
            }
        }
    }
}

// ---------------------------------------------------------------------------
// Flash attention: one wave per 16-row query tile, 32-key blocks.
// Qp,Kp: (b,h,t,hd) f16.  VpT: (b,h,hd,t) f16.  Oout: (b,t,c) f16.
// ---------------------------------------------------------------------------
__global__ void __launch_bounds__(128)
attn_wmma(const _Float16* __restrict__ Qp, const _Float16* __restrict__ Kp,
          const _Float16* __restrict__ VpT, _Float16* __restrict__ Oout) {
    __shared__ __align__(16) _Float16 Plds[4][16][40];

    const int tid  = threadIdx.x;
    const int lane = tid & 31;
    const int wid  = tid >> 5;
    const int lm   = lane & 15;
    const int lh   = lane >> 4;
    const int bh   = blockIdx.y;
    const int b    = bh >> 4, h = bh & 15;
    const int i0   = (blockIdx.x * 4 + wid) * 16;

    const float scale = 0.03125f;     // 1/sqrt(1024)
    const float NEGB  = -1e30f;

    const _Float16* qrow = Qp + ((size_t)bh * TT + i0 + lm) * HD;
    const int klo = lh * 8;
    v16h qf0 = cat8(*(const v8h*)(qrow + lh * 16),      *(const v8h*)(qrow + lh * 16 + 8));
    v16h qf1 = cat8(*(const v8h*)(qrow + 32 + lh * 16), *(const v8h*)(qrow + 40 + lh * 16));

    v8f o[4] = {};
    float m_run[8], l_run[8];
#pragma unroll
    for (int r = 0; r < 8; ++r) { m_run[r] = NEGB; l_run[r] = 0.0f; }

    const int num_kb = (i0 + 15) / 32 + 1;
    for (int kb = 0; kb < num_kb; ++kb) {
        const int j0 = kb * 32;

        const _Float16* krow0 = Kp + ((size_t)bh * TT + j0 + lm) * HD + lh * 16;
        const _Float16* krow1 = krow0 + 16 * HD;
        v16h kf00 = cat8(*(const v8h*)krow0,        *(const v8h*)(krow0 + 8));
        v16h kf01 = cat8(*(const v8h*)(krow0 + 32), *(const v8h*)(krow0 + 40));
        v16h kf10 = cat8(*(const v8h*)krow1,        *(const v8h*)(krow1 + 8));
        v16h kf11 = cat8(*(const v8h*)(krow1 + 32), *(const v8h*)(krow1 + 40));

        v8f s0 = {}, s1 = {};
        s0 = wmma_f16(qf0, kf00, s0);
        s0 = wmma_f16(qf1, kf01, s0);
        s1 = wmma_f16(qf0, kf10, s1);
        s1 = wmma_f16(qf1, kf11, s1);

#pragma unroll
        for (int r = 0; r < 8; ++r) {
            const int irow = i0 + r + lh * 8;
            const int ja = j0 + lm, jb = j0 + lm + 16;
            float sa = (ja <= irow) ? s0[r] * scale : NEGB;
            float sb = (jb <= irow) ? s1[r] * scale : NEGB;
            float rm = fmaxf(sa, sb);
#pragma unroll
            for (int mm = 8; mm >= 1; mm >>= 1)
                rm = fmaxf(rm, __shfl_xor(rm, mm, 32));
            const float mnew  = fmaxf(m_run[r], rm);
            const float alpha = __expf(m_run[r] - mnew);
            m_run[r] = mnew;
            const float pa = __expf(sa - mnew);
            const float pb = __expf(sb - mnew);
            float rs = pa + pb;
#pragma unroll
            for (int mm = 8; mm >= 1; mm >>= 1)
                rs += __shfl_xor(rs, mm, 32);
            l_run[r] = l_run[r] * alpha + rs;
#pragma unroll
            for (int d = 0; d < 4; ++d) o[d][r] *= alpha;
            const int mrow = r + lh * 8;
            Plds[wid][mrow][lm]      = (_Float16)pa;
            Plds[wid][mrow][lm + 16] = (_Float16)pb;
        }
        asm volatile("s_wait_dscnt 0" ::: "memory");  // LDS in-order per wave

        v16h pf = cat8(*(const v8h*)&Plds[wid][lm][lh * 16],
                       *(const v8h*)&Plds[wid][lm][lh * 16 + 8]);

#pragma unroll
        for (int d = 0; d < 4; ++d) {
            const _Float16* vrow =
                VpT + ((size_t)bh * HD + d * 16 + lm) * TT + j0 + lh * 16;
            v16h vf = cat8(*(const v8h*)vrow, *(const v8h*)(vrow + 8));
            o[d] = wmma_f16(pf, vf, o[d]);
        }
    }

#pragma unroll
    for (int d = 0; d < 4; ++d) {
#pragma unroll
        for (int r = 0; r < 8; ++r) {
            const float val = o[d][r] / l_run[r];
            const size_t idx =
                ((size_t)b * TT + i0 + r + lh * 8) * CC + h * HD + d * 16 + lm;
            Oout[idx] = (_Float16)val;
        }
    }
}

// ---------------------------------------------------------------------------
extern "C" void kernel_launch(void* const* d_in, const int* in_sizes, int n_in,
                              void* d_out, int out_size, void* d_ws, size_t ws_size,
                              hipStream_t stream) {
    (void)in_sizes; (void)n_in; (void)out_size;
    const float* q  = (const float*)d_in[0];
    const float* k  = (const float*)d_in[1];
    const float* v  = (const float*)d_in[2];
    const float* Wq = (const float*)d_in[3];
    const float* bq = (const float*)d_in[4];
    const float* Wk = (const float*)d_in[5];
    const float* bk = (const float*)d_in[6];
    const float* Wv = (const float*)d_in[7];
    const float* bv = (const float*)d_in[8];
    const float* Wo = (const float*)d_in[9];
    const float* bo = (const float*)d_in[10];

    const size_t act  = (size_t)MROWS * CC;            // 8,388,608 elems
    const size_t actB = act * sizeof(_Float16);        // 16 MB
    const size_t wgt  = (size_t)CC * CC;               // 1,048,576 elems
    const size_t wgtB = wgt * sizeof(_Float16);        // 2 MB
    if (ws_size < 7 * actB + 4 * wgtB) return;         // 120 MB scratch

    char* ws = (char*)d_ws;
    _Float16* Qp   = (_Float16*)(ws + 0 * actB);
    _Float16* Kp   = (_Float16*)(ws + 1 * actB);
    _Float16* VpT  = (_Float16*)(ws + 2 * actB);
    _Float16* AOut = (_Float16*)(ws + 3 * actB);
    _Float16* Qh   = (_Float16*)(ws + 4 * actB);
    _Float16* Kh   = (_Float16*)(ws + 5 * actB);
    _Float16* Vh   = (_Float16*)(ws + 6 * actB);
    _Float16* Wqh  = (_Float16*)(ws + 7 * actB + 0 * wgtB);
    _Float16* Wkh  = (_Float16*)(ws + 7 * actB + 1 * wgtB);
    _Float16* Wvh  = (_Float16*)(ws + 7 * actB + 2 * wgtB);
    _Float16* Woh  = (_Float16*)(ws + 7 * actB + 3 * wgtB);

    const int actG = (int)(act / 4 / 256);   // 8192 blocks
    const int wgtG = (int)(wgt / 4 / 256);   // 1024 blocks
    cvt_f16<<<actG, 256, 0, stream>>>(q,  Qh,  (int)(act / 4));
    cvt_f16<<<actG, 256, 0, stream>>>(k,  Kh,  (int)(act / 4));
    cvt_f16<<<actG, 256, 0, stream>>>(v,  Vh,  (int)(act / 4));
    cvt_f16<<<wgtG, 256, 0, stream>>>(Wq, Wqh, (int)(wgt / 4));
    cvt_f16<<<wgtG, 256, 0, stream>>>(Wk, Wkh, (int)(wgt / 4));
    cvt_f16<<<wgtG, 256, 0, stream>>>(Wv, Wvh, (int)(wgt / 4));
    cvt_f16<<<wgtG, 256, 0, stream>>>(Wo, Woh, (int)(wgt / 4));

    dim3 gg(MROWS / 128, CC / 128);   // (64, 8)
    gemm128_wmma<<<gg, 256, 0, stream>>>(Qh, Wqh, bq, Qp,  0);
    gemm128_wmma<<<gg, 256, 0, stream>>>(Kh, Wkh, bk, Kp,  0);
    gemm128_wmma<<<gg, 256, 0, stream>>>(Vh, Wvh, bv, VpT, 1);

    attn_wmma<<<dim3(TT / 64, BB * NH), 128, 0, stream>>>(Qp, Kp, VpT, AOut);

    gemm128_wmma<<<gg, 256, 0, stream>>>(AOut, Woh, bo, d_out, 2);
}
